// NeuralWDRC_66305705116231
// MI455X (gfx1250) — compile-verified, matrix operand order checked by reading, not run on hardware
//
#include <hip/hip_runtime.h>
#include <hip/hip_bf16.h>
#include <math.h>

typedef __attribute__((ext_vector_type(16))) _Float16 v16h;
typedef __attribute__((ext_vector_type(8)))  _Float16 v8h;
typedef __attribute__((ext_vector_type(8)))  float    v8f;

#define GRU_HD 256
#define H1D    128
#define H2D    64
#define BATCH  16
#define TLEN   4000
#define NLEN   320000
#define CCHUNK 2048
#define CWARM  512

// ---- gfx1250 async copy to LDS (ASYNCcnt-tracked) -------------------------
// lds_off: wave-relative LDS byte offset (= low 32 bits of a generic pointer
// to a __shared__ object); gsrc: 64-bit global address in a VGPR pair.
__device__ __forceinline__ void async_copy_b128(unsigned lds_off,
                                                const float* gsrc) {
    asm volatile("global_load_async_to_lds_b128 %0, %1, off"
                 :: "v"(lds_off), "v"(gsrc)
                 : "memory");
}

#if defined(__has_builtin)
# if __has_builtin(__builtin_amdgcn_s_wait_asynccnt)
#  define WAIT_ASYNC0() __builtin_amdgcn_s_wait_asynccnt(0)
# endif
#endif
#ifndef WAIT_ASYNC0
# define WAIT_ASYNC0() asm volatile("s_wait_asynccnt 0" ::: "memory")
#endif

// Load a 16-element f16 WMMA fragment (A row-major per lane / B col-major per
// lane, identical K indexing) from fp32 memory (global or LDS). `base`
// already includes the per-lane row offset and the half*8 K offset; elements
// 0..7 are 8 contiguous floats at base, elements 8..15 at base+16.
__device__ __forceinline__ v16h load_frag_f32(const float* base) {
    const float4* p0 = (const float4*)(base);
    const float4* p1 = (const float4*)(base + 16);
    float4 x0 = p0[0], x1 = p0[1];
    float4 x2 = p1[0], x3 = p1[1];
    v16h f;
    f[0]  = (_Float16)x0.x; f[1]  = (_Float16)x0.y; f[2]  = (_Float16)x0.z; f[3]  = (_Float16)x0.w;
    f[4]  = (_Float16)x1.x; f[5]  = (_Float16)x1.y; f[6]  = (_Float16)x1.z; f[7]  = (_Float16)x1.w;
    f[8]  = (_Float16)x2.x; f[9]  = (_Float16)x2.y; f[10] = (_Float16)x2.z; f[11] = (_Float16)x2.w;
    f[12] = (_Float16)x3.x; f[13] = (_Float16)x3.y; f[14] = (_Float16)x3.z; f[15] = (_Float16)x3.w;
    return f;
}

// Fused 3-layer MLP over a 16-row tile; emits only the compressor ratio.
// Stage: async-copy the 16x256 fp32 gru tile to LDS once per block (no 8x
// redundant global reads). 8 waves: layer1 -> each wave one 16-col tile of
// 128; layer2 -> waves 0..3; layer3 (64->3) on wave 0 lanes 0..15.
__global__ __launch_bounds__(256)
void wdrc_mlp_kernel(const float* __restrict__ gru,
                     const float* __restrict__ W1, const float* __restrict__ b1,
                     const float* __restrict__ a1p,
                     const float* __restrict__ W2, const float* __restrict__ b2,
                     const float* __restrict__ a2p,
                     const float* __restrict__ W3, const float* __restrict__ b3,
                     float* __restrict__ ratio_out)
{
    __shared__ float     af32[16 * GRU_HD];   // staged A tile (16 KB)
    __shared__ _Float16  h1s[16 * H1D];
    __shared__ _Float16  h2s[16 * H2D];

    const int lane  = threadIdx.x & 31;
    const int wave  = threadIdx.x >> 5;
    const int rBase = blockIdx.x * 16;
    const int m     = lane & 15;   // M index (A) / N index (B): same lane math
    const int hf    = lane >> 4;   // lane half selects K sub-ranges
    const float a1  = a1p[0];
    const float a2  = a2p[0];

    // ---------------- async stage: gru tile -> LDS -------------------------
    {
        const unsigned lds_base = (unsigned)(unsigned long long)(void*)af32;
        #pragma unroll
        for (int j = 0; j < 4; ++j) {
            const int c   = (int)threadIdx.x + 256 * j;  // b128 chunk id, 0..1023
            const int row = c >> 6;                      // 64 chunks per row
            const int col = (c & 63) << 2;               // float index in row
            async_copy_b128(lds_base + (unsigned)c * 16u,
                            gru + (size_t)(rBase + row) * GRU_HD + col);
        }
        // warm the W2 row this wave will need in layer 2 (global_prefetch_b8)
        if (wave < 4) __builtin_prefetch(W2 + (size_t)(wave * 16 + m) * H1D, 0, 1);
        WAIT_ASYNC0();        // my wave's async transfers have landed in LDS
        __syncthreads();      // everyone's have
    }

    // ---------------- layer 1: (16x256) x (256x128) via wmma f16 ----------
    {
        const int cBase = wave * 16;
        const float* arow = af32 + m * GRU_HD + hf * 8;                    // LDS
        const float* brow = W1 + (size_t)(cBase + m) * GRU_HD + hf * 8;    // global
        v8f acc = {};
        #pragma unroll
        for (int k0 = 0; k0 < GRU_HD; k0 += 32) {
            v16h a = load_frag_f32(arow + k0);
            v16h b = load_frag_f32(brow + k0);
            acc = __builtin_amdgcn_wmma_f32_16x16x32_f16(
                      false, a, false, b, (short)0, acc, false, false);
        }
        #pragma unroll
        for (int i = 0; i < 8; ++i) {
            const int r = i + 8 * hf;           // C layout: M = vgpr + 8*half
            const int c = cBase + m;            // N = lane & 15
            float v = acc[i] + b1[c];
            v = (v >= 0.0f) ? v : a1 * v;       // PReLU
            h1s[r * H1D + c] = (_Float16)v;
        }
    }
    __syncthreads();

    // ---------------- layer 2: (16x128) x (128x64), waves 0..3 ------------
    if (wave < 4) {
        const int cBase = wave * 16;
        const float* brow = W2 + (size_t)(cBase + m) * H1D + hf * 8;
        v8f acc = {};
        #pragma unroll
        for (int k0 = 0; k0 < H1D; k0 += 32) {
            v8h lo = *(const v8h*)&h1s[m * H1D + k0 + hf * 8];
            v8h hi = *(const v8h*)&h1s[m * H1D + k0 + 16 + hf * 8];
            v16h a;
            #pragma unroll
            for (int e = 0; e < 8; ++e) { a[e] = lo[e]; a[e + 8] = hi[e]; }
            v16h b = load_frag_f32(brow + k0);
            acc = __builtin_amdgcn_wmma_f32_16x16x32_f16(
                      false, a, false, b, (short)0, acc, false, false);
        }
        #pragma unroll
        for (int i = 0; i < 8; ++i) {
            const int r = i + 8 * hf;
            const int c = cBase + m;
            float v = acc[i] + b2[c];
            v = (v >= 0.0f) ? v : a2 * v;
            h2s[r * H2D + c] = (_Float16)v;
        }
    }
    __syncthreads();

    // ---------------- layer 3: only p[2] -> ratio survives downstream -----
    if (wave == 0 && lane < 16) {
        float p2 = b3[2];
        #pragma unroll 8
        for (int k = 0; k < H2D; ++k)
            p2 += (float)h2s[lane * H2D + k] * W3[2 * H2D + k];
        float sp = (p2 > 20.0f) ? p2 : log1pf(expf(p2));      // softplus
        float ratio = fminf(fmaxf(sp + 1.0f, 1.0f), 20.0f);
        ratio_out[rBase + lane] = ratio;
    }
}

__global__ void wdrc_init_kernel(unsigned int* __restrict__ maxbits) {
    if (threadIdx.x < 2) maxbits[threadIdx.x] = 0u;
}

// Chunked dual-IIR compressor. One wave per (batch, 2048-sample chunk);
// 512-sample warm-up makes chunks independent (0.9^512 << fp32 ulp).
// Within the wave: 5-step Kogge-Stone scan of the constant-decay recurrence.
__global__ __launch_bounds__(32)
void wdrc_compress_kernel(const float* __restrict__ enhanced,
                          const float* __restrict__ noisy,
                          const float* __restrict__ ratio,
                          float* __restrict__ out,
                          unsigned int* __restrict__ maxbits)
{
    const int b     = blockIdx.y;
    const int chunk = blockIdx.x;
    const int lane  = threadIdx.x;
    const int cStart = chunk * CCHUNK;
    const int cEnd   = min(cStart + CCHUNK, NLEN);
    if (cStart >= NLEN) return;

    const float* enh = enhanced + (size_t)b * NLEN;
    const float* noi = noisy    + (size_t)b * NLEN;
    const float* rat = ratio    + (size_t)b * TLEN;
    float*       ob  = out      + (size_t)b * NLEN;

    auto gains = [&](int t, float& e, float& r, float& gE, float& gR) {
        e = enh[t];
        r = noi[t] - e;
        float src = (t + 0.5f) * ((float)TLEN / (float)NLEN) - 0.5f;
        src = fminf(fmaxf(src, 0.0f), (float)(TLEN - 1));
        int   i0 = (int)src;                   // src >= 0 -> trunc == floor
        int   i1 = min(i0 + 1, TLEN - 1);
        float w  = src - (float)i0;
        float rt = rat[i0] * (1.0f - w) + rat[i1] * w;
        float envE = fabsf(e);
        float grE  = (envE > 0.3f) ? 0.3f + (envE - 0.3f) / rt : envE;
        gE = fminf(fmaxf(grE / (envE + 1e-8f), 0.1f), 2.0f);
        float rtR  = 0.5f * rt;
        float envR = fabsf(r);
        float grR  = (envR > 0.1f) ? 0.1f + (envR - 0.1f) / rtR : envR;
        gR = fminf(fmaxf(grR / (envR + 1e-8f), 0.1f), 2.0f);
    };

    float carryE, carryR;
    int   tBegin;
    float maxO = 0.0f, maxE = 0.0f;

    if (chunk == 0) {
        float e0, r0, gE0, gR0;
        gains(0, e0, r0, gE0, gR0);
        carryE = gE0; carryR = gR0;            // s_0 = g_0 per reference
        if (lane == 0) {
            float o = e0 * carryE + 0.1f * r0 * carryR;
            ob[0] = o;
            maxO = fabsf(o); maxE = fabsf(e0);
        }
        tBegin = 1;
    } else {
        tBegin = cStart - CWARM;
        float e0, r0, gE0, gR0;
        gains(tBegin, e0, r0, gE0, gR0);
        carryE = gE0; carryR = gR0;            // any seed: decays to 0 by cStart
    }

    const float dl = powf(0.9f, (float)(lane + 1));   // per-lane carry decay

    for (int base = tBegin; base < cEnd; base += 32) {
        int t  = base + lane;
        int tc = min(t, NLEN - 1);
        float e, r, gE, gR;
        gains(tc, e, r, gE, gR);
        float vE = 0.1f * gE, vR = 0.1f * gR;
        float u;
        u = __shfl_up(vE, 1);  if (lane >= 1)  vE += 0.9f * u;
        u = __shfl_up(vR, 1);  if (lane >= 1)  vR += 0.9f * u;
        u = __shfl_up(vE, 2);  if (lane >= 2)  vE += 0.81f * u;
        u = __shfl_up(vR, 2);  if (lane >= 2)  vR += 0.81f * u;
        u = __shfl_up(vE, 4);  if (lane >= 4)  vE += 0.6561f * u;
        u = __shfl_up(vR, 4);  if (lane >= 4)  vR += 0.6561f * u;
        u = __shfl_up(vE, 8);  if (lane >= 8)  vE += 0.43046721f * u;
        u = __shfl_up(vR, 8);  if (lane >= 8)  vR += 0.43046721f * u;
        u = __shfl_up(vE, 16); if (lane >= 16) vE += 0.18530201888518410f * u;
        u = __shfl_up(vR, 16); if (lane >= 16) vR += 0.18530201888518410f * u;
        float sE = vE + dl * carryE;
        float sR = vR + dl * carryR;
        if (t >= cStart && t < cEnd) {
            float o = e * sE + 0.1f * r * sR;
            ob[t] = o;
            maxO = fmaxf(maxO, fabsf(o));
            maxE = fmaxf(maxE, fabsf(e));
        }
        carryE = __shfl(sE, 31);
        carryR = __shfl(sR, 31);
    }

    #pragma unroll
    for (int d = 16; d >= 1; d >>= 1) {
        maxO = fmaxf(maxO, __shfl_xor(maxO, d));
        maxE = fmaxf(maxE, __shfl_xor(maxE, d));
    }
    if (lane == 0) {
        atomicMax(&maxbits[0], __float_as_uint(maxO));   // non-neg floats order as u32
        atomicMax(&maxbits[1], __float_as_uint(maxE));
    }
}

__global__ void wdrc_norm_kernel(float* __restrict__ out,
                                 const unsigned int* __restrict__ maxbits,
                                 int n)
{
    const float mo = __uint_as_float(maxbits[0]);
    const float me = __uint_as_float(maxbits[1]);
    const float s  = me / (mo + 1e-8f);
    int i = blockIdx.x * blockDim.x + threadIdx.x;
    const int stride = gridDim.x * blockDim.x;
    for (; i < n; i += stride) out[i] *= s;
}

extern "C" void kernel_launch(void* const* d_in, const int* in_sizes, int n_in,
                              void* d_out, int out_size, void* d_ws, size_t ws_size,
                              hipStream_t stream)
{
    (void)in_sizes; (void)n_in; (void)ws_size;
    const float* gru      = (const float*)d_in[0];
    const float* enhanced = (const float*)d_in[1];
    const float* noisy    = (const float*)d_in[2];
    const float* W1 = (const float*)d_in[3];
    const float* b1 = (const float*)d_in[4];
    const float* a1 = (const float*)d_in[5];
    const float* W2 = (const float*)d_in[6];
    const float* b2 = (const float*)d_in[7];
    const float* a2 = (const float*)d_in[8];
    const float* W3 = (const float*)d_in[9];
    const float* b3 = (const float*)d_in[10];

    float* out = (float*)d_out;
    unsigned int* maxbits = (unsigned int*)d_ws;
    float* ratio = (float*)((char*)d_ws + 256);   // B*T floats of scratch

    wdrc_init_kernel<<<1, 32, 0, stream>>>(maxbits);
    wdrc_mlp_kernel<<<(BATCH * TLEN) / 16, 256, 0, stream>>>(
        gru, W1, b1, a1, W2, b2, a2, W3, b3, ratio);
    dim3 cg((NLEN + CCHUNK - 1) / CCHUNK, BATCH);
    wdrc_compress_kernel<<<cg, 32, 0, stream>>>(enhanced, noisy, ratio, out, maxbits);
    wdrc_norm_kernel<<<512, 256, 0, stream>>>(out, maxbits, out_size);
}
